// MultiScaleFeatureExtractor_74912819577594
// MI455X (gfx1250) — matrix operand order repaired and checked by reading, not compiled, not run on hardware
//
#include <hip/hip_runtime.h>

// ---------------------------------------------------------------------------
// MI455X fused slice-attention aggregation.
//   B=2, N=100000, C=256, H=8, D=64, S=64, INNER=512
// ~1.3e11 FLOPs vs 205MB mandatory x-read. Fully fused per 64-token tile:
//   bf16 WMMA (16x16x32) for all three GEMMs, persistent register accumulators,
//   double-buffered async global->LDS staging of x (ASYNCcnt path).
// ---------------------------------------------------------------------------

typedef __attribute__((ext_vector_type(16))) __bf16 v16bf;
typedef __attribute__((ext_vector_type(8)))  float  v8f;

#define B_     2
#define N_     100000
#define C_     256
#define H_     8
#define D_     64
#define S_     64
#define INNER_ 512
#define TM     64
#define NTILES ((N_ + TM - 1) / TM)   // 1563

// --- LDS layout (bytes), 288KB total of 320KB/WGP ---
#define OFF_PXB   0           // 64KB: px bf16 [TM][INNER]; reused as swT bf16 [H*S][TM]
#define OFF_FXBT  (64*1024)   // 64KB: fx^T bf16 [INNER][TM]
#define OFF_SP    (128*1024)  // 128KB: slice_proj f32 [TM][H][S]; first 64KB doubles
                              //        as async staging buffer for raw f32 x tile
#define OFF_XB    (256*1024)  // 32KB: x tile bf16 [TM][C]
#define LDS_BYTES (288*1024)

// --- workspace layout (bytes) ---
#define WS_ACC    0                          // f32 [B][H][S][D]
#define WS_NORM   (B_*H_*S_*D_*4)            // f32 [B][H][S]
#define WS_WCAT   (WS_NORM + B_*H_*S_*4)     // bf16 swizzled [64 nt][8 ks][32 lane][16]
#define WS_WSLICE (WS_WCAT + 64*8*32*16*2)   // bf16 swizzled [4 nt][2 ks][32 lane][16]

// A-fragment (16x32 bf16): lane l -> row = l&15, K = kb..kb+15, kb=(l>>4)*16
__device__ __forceinline__ v16bf load_A(const __bf16* base, int ld, int lane) {
  const int r  = lane & 15;
  const int kb = (lane >> 4) * 16;
  const __bf16* p = base + r * ld + kb;
  union { v16bf v; uint4 q[2]; } f;
  f.q[0] = *(const uint4*)(p);
  f.q[1] = *(const uint4*)(p + 8);
  return f.v;
}

// B-fragment: 16 contiguous bf16 (pre-swizzled / transposed layouts)
__device__ __forceinline__ v16bf load_B16(const __bf16* p) {
  union { v16bf v; uint4 q[2]; } f;
  f.q[0] = ((const uint4*)p)[0];
  f.q[1] = ((const uint4*)p)[1];
  return f.v;
}

// Async DMA 16B global -> LDS (GLOBAL_LOAD_ASYNC_TO_LDS_B128, ASYNCcnt).
__device__ __forceinline__ void async_ld_b128(unsigned lds_addr, const void* gptr) {
  asm volatile("global_load_async_to_lds_b128 %0, %1, off"
               :: "v"(lds_addr), "v"(gptr) : "memory");
}
__device__ __forceinline__ void wait_async0() {
  asm volatile("s_wait_asynccnt 0x0" ::: "memory");
}

// ---------------------------------------------------------------------------
// Prep: zero accumulators, pre-swizzle weights into WMMA B-fragment order.
// B-fragment layout: lane l -> col n0+(l&15), K = (l>>4)*16 + e.
// ---------------------------------------------------------------------------
__global__ void k_prep(const float* __restrict__ Wx, const float* __restrict__ Wfx,
                       const float* __restrict__ Wsl,
                       float* __restrict__ acc, float* __restrict__ nrm,
                       __bf16* __restrict__ wcat, __bf16* __restrict__ wslice) {
  const int tid = blockIdx.x * blockDim.x + threadIdx.x;
  const int stride = gridDim.x * blockDim.x;
  for (int i = tid; i < B_*H_*S_*D_; i += stride) acc[i] = 0.0f;
  for (int i = tid; i < B_*H_*S_;    i += stride) nrm[i] = 0.0f;
  for (int f = tid; f < 64*8*32*16; f += stride) {   // Wcat = [W_x | W_fx] (C,1024)
    const int e  = f & 15;
    const int l  = (f >> 4) & 31;
    const int ks = (f >> 9) & 7;
    const int nt = f >> 12;
    const int k  = ks*32 + (l >> 4)*16 + e;
    const int n  = nt*16 + (l & 15);
    const float v = (n < INNER_) ? Wx[k*INNER_ + n] : Wfx[k*INNER_ + (n - INNER_)];
    wcat[f] = (__bf16)v;
  }
  for (int f = tid; f < 4*2*32*16; f += stride) {    // W_slice (D,S)
    const int e  = f & 15;
    const int l  = (f >> 4) & 31;
    const int ks = (f >> 9) & 1;
    const int nt = f >> 10;
    const int k  = ks*32 + (l >> 4)*16 + e;
    const int n  = nt*16 + (l & 15);
    wslice[f] = (__bf16)Wsl[k*S_ + n];
  }
}

// ---------------------------------------------------------------------------
// Main fused kernel. 256 threads = 8 waves (wave32). Persistent over tiles.
// ---------------------------------------------------------------------------
__global__ void __launch_bounds__(256, 1)
k_main(const float* __restrict__ x,
       const float* __restrict__ bx, const float* __restrict__ bfx,
       const float* __restrict__ bsl, const float* __restrict__ temp,
       const __bf16* __restrict__ wcat, const __bf16* __restrict__ wslice,
       float* __restrict__ g_acc, float* __restrict__ g_norm) {
  extern __shared__ __align__(16) char smem[];
  __bf16* pxb  = (__bf16*)(smem + OFF_PXB);   // [TM][INNER]
  __bf16* swT  = (__bf16*)(smem + OFF_PXB);   // alias: [H*S][TM] (after GEMM2)
  __bf16* fxbT = (__bf16*)(smem + OFF_FXBT);  // [INNER][TM]
  float*  sp   = (float* )(smem + OFF_SP);    // [TM][H][S]
  float*  spf  = (float* )(smem + OFF_SP);    // staging view: f32 [TM][C]
  __bf16* xb   = (__bf16*)(smem + OFF_XB);    // [TM][C]

  const int tid  = threadIdx.x;
  const int lane = tid & 31;
  const int wave = tid >> 5;
  const int b    = blockIdx.y;
  const int nl   = lane & 15;
  const int mb   = (lane >> 4) * 8;

  // persistent GEMM3 accumulators: wave == head; 16 tiles (4 s-tiles x 4 d-tiles)
  v8f acc3[16];
#pragma unroll
  for (int i = 0; i < 16; ++i) acc3[i] = (v8f){0,0,0,0,0,0,0,0};
  float nrm0 = 0.0f, nrm1 = 0.0f;

  const float* xB = x + (size_t)b * N_ * C_;
  const unsigned lds_spf = (unsigned)(uintptr_t)spf;   // LDS byte offset (addr[31:0])

  // ---- prologue: async-stage first x tile into spf ----
  {
    const int t0 = blockIdx.x * TM;
    for (int c = tid; c < TM * (C_/4); c += 256) {
      const int row = c >> 6;                          // 64 float4 per row
      if (t0 + row < N_)
        async_ld_b128(lds_spf + (unsigned)c * 16u,
                      xB + (size_t)(t0 + row) * C_ + (c & 63) * 4);
    }
  }

  for (int tile = blockIdx.x; tile < NTILES; tile += gridDim.x) {
    const int t0 = tile * TM;

    // ---- wait for staged tile, convert f32 -> bf16 into xb ----
    wait_async0();
    __syncthreads();
    for (int i = tid; i < TM * (C_/4); i += 256) {
      const int row = i >> 6;
      float4 v = ((const float4*)spf)[i];
      if (t0 + row >= N_) v = (float4){0.f, 0.f, 0.f, 0.f};
      union { __bf16 h[4]; uint2 u; } o;
      o.h[0] = (__bf16)v.x; o.h[1] = (__bf16)v.y;
      o.h[2] = (__bf16)v.z; o.h[3] = (__bf16)v.w;
      *(uint2*)(xb + row * C_ + (i & 63) * 4) = o.u;
    }
    __syncthreads();

    // ---- GEMM1: x(64x256) @ Wcat(256x1024) -> px, fx^T ----
    for (int j = 0; j < 32; ++j) {
      const int idx = wave + 8*j;        // 0..255
      const int mt  = idx >> 6;          // token tile 0..3
      const int ntg = idx & 63;          // output col tile 0..63
      v8f acc = (v8f){0,0,0,0,0,0,0,0};
      const __bf16* wbase = wcat + (size_t)ntg * (8*32*16);
#pragma unroll
      for (int ks = 0; ks < 8; ++ks) {
        const v16bf a  = load_A(xb + mt*16*C_ + ks*32, C_, lane);
        const v16bf bb = load_B16(wbase + (ks*32 + lane)*16);
        acc = __builtin_amdgcn_wmma_f32_16x16x32_bf16(false, a, false, bb, (short)0, acc, false, false);
      }
      if (ntg < 32) {                    // px -> token-major
        const int col  = ntg*16 + nl;
        const float bias = bx[col];
#pragma unroll
        for (int r = 0; r < 8; ++r)
          pxb[(mt*16 + mb + r) * INNER_ + col] = (__bf16)(acc[r] + bias);
      } else {                           // fx -> feature-major (transposed)
        const int fcol = (ntg - 32)*16 + nl;
        const float bias = bfx[fcol];
#pragma unroll
        for (int r = 0; r < 8; ++r)
          fxbT[fcol * TM + (mt*16 + mb + r)] = (__bf16)(acc[r] + bias);
      }
    }
    __syncthreads();

    // ---- GEMM2: per-head px(64x64) @ Wslice(64x64) -> logits f32 ----
    for (int j = 0; j < 16; ++j) {
      const int idx = wave + 8*j;        // 0..127 = mt(4) x h(8) x st(4)
      const int st  = idx & 3;
      const int h   = (idx >> 2) & 7;
      const int mt  = idx >> 5;
      v8f acc = (v8f){0,0,0,0,0,0,0,0};
#pragma unroll
      for (int ks = 0; ks < 2; ++ks) {
        const v16bf a  = load_A(pxb + (mt*16)*INNER_ + h*D_ + ks*32, INNER_, lane);
        const v16bf bb = load_B16(wslice + ((st*2 + ks)*32 + lane)*16);
        acc = __builtin_amdgcn_wmma_f32_16x16x32_bf16(false, a, false, bb, (short)0, acc, false, false);
      }
      const int s = st*16 + nl;
      const float bias = bsl[s];
#pragma unroll
      for (int r = 0; r < 8; ++r)
        sp[((mt*16 + mb + r)*H_ + h)*S_ + s] = acc[r] + bias;
    }
    __syncthreads();

    // ---- softmax over S per (token,head); write bf16 weights transposed ----
#pragma unroll
    for (int rr = 0; rr < 2; ++rr) {
      const int row = tid + rr*256;      // row = t*H + h
      const int t   = row >> 3;
      const int h   = row & 7;
      float tc = temp[h];
      tc = fminf(fmaxf(tc, 0.1f), 5.0f);
      const float inv = 1.0f / tc;
      float4* pr4 = (float4*)(sp + row * S_);
      float m = -1e30f;
      for (int i = 0; i < S_/4; ++i) {
        const float4 v = pr4[i];
        m = fmaxf(m, fmaxf(fmaxf(v.x, v.y), fmaxf(v.z, v.w)) * inv);
      }
      float sum = 0.0f;
      for (int i = 0; i < S_/4; ++i) {
        float4 v = pr4[i];
        v.x = __expf(v.x*inv - m); v.y = __expf(v.y*inv - m);
        v.z = __expf(v.z*inv - m); v.w = __expf(v.w*inv - m);
        sum += v.x + v.y + v.z + v.w;
        pr4[i] = v;
      }
      const float rs = ((t0 + t) < N_) ? (1.0f / sum) : 0.0f;  // mask padded tokens
      const float* pr = sp + row * S_;
      for (int s = 0; s < S_; ++s)
        swT[(h*S_ + s)*TM + t] = (__bf16)(pr[s] * rs);
    }
    __syncthreads();

    // ---- kick off async staging of next tile into spf (sp is dead now);
    //      DMA overlaps the norm reduction + GEMM3 below ----
    {
      const int ntile = tile + gridDim.x;
      if (ntile < NTILES) {
        const int nt0 = ntile * TM;
        for (int c = tid; c < TM * (C_/4); c += 256) {
          const int row = c >> 6;
          if (nt0 + row < N_)
            async_ld_b128(lds_spf + (unsigned)c * 16u,
                          xB + (size_t)(nt0 + row) * C_ + (c & 63) * 4);
        }
      }
    }

    // ---- norm partials: sum over tokens of each (h,s) weight row ----
    {
      const uint4* r0 = (const uint4*)(swT + tid * TM);
      const uint4* r1 = (const uint4*)(swT + (tid + 256) * TM);
      float a0 = 0.0f, a1 = 0.0f;
#pragma unroll
      for (int q = 0; q < TM/8; ++q) {
        union { uint4 u; __bf16 h[8]; } w0, w1;
        w0.u = r0[q]; w1.u = r1[q];
#pragma unroll
        for (int e = 0; e < 8; ++e) { a0 += (float)w0.h[e]; a1 += (float)w1.h[e]; }
      }
      nrm0 += a0; nrm1 += a1;
    }

    // ---- GEMM3: per-head w^T(64x64) @ fx(64x64), accumulate in registers ----
    {
      const int h = wave;
#pragma unroll
      for (int i = 0; i < 16; ++i) {
        const int st = i >> 2, dt = i & 3;
#pragma unroll
        for (int ks = 0; ks < 2; ++ks) {
          const v16bf a  = load_A(swT + (h*S_ + st*16)*TM + ks*32, TM, lane);
          const v16bf bb = load_B16(fxbT + (h*D_ + dt*16 + nl)*TM + ks*32 + (lane >> 4)*16);
          acc3[i] = __builtin_amdgcn_wmma_f32_16x16x32_bf16(false, a, false, bb, (short)0, acc3[i], false, false);
        }
      }
    }
    __syncthreads();   // protect swT(=pxb) before next tile's GEMM1
  }

  // ---- flush persistent accumulators (L2-resident atomics) ----
  atomicAdd(&g_norm[b*H_*S_ + tid],       nrm0);
  atomicAdd(&g_norm[b*H_*S_ + tid + 256], nrm1);
  {
    const int h = wave;
#pragma unroll
    for (int i = 0; i < 16; ++i) {
      const int st = i >> 2, dt = i & 3;
#pragma unroll
      for (int r = 0; r < 8; ++r) {
        const int s = st*16 + mb + r;
        const int d = dt*16 + nl;
        atomicAdd(&g_acc[(((size_t)b*H_ + h)*S_ + s)*D_ + d], acc3[i][r]);
      }
    }
  }
}

// ---------------------------------------------------------------------------
__global__ void k_final(const float* __restrict__ acc, const float* __restrict__ nrm,
                        float* __restrict__ out) {
  const int i = blockIdx.x * 256 + threadIdx.x;
  if (i < B_*H_*S_*D_) out[i] = acc[i] / (nrm[i / D_] + 1e-5f);
}

// ---------------------------------------------------------------------------
extern "C" void kernel_launch(void* const* d_in, const int* in_sizes, int n_in,
                              void* d_out, int out_size, void* d_ws, size_t ws_size,
                              hipStream_t stream) {
  const float* x    = (const float*)d_in[0];
  const float* Wx   = (const float*)d_in[1];
  const float* bx   = (const float*)d_in[2];
  const float* Wfx  = (const float*)d_in[3];
  const float* bfx  = (const float*)d_in[4];
  const float* Wsl  = (const float*)d_in[5];
  const float* bsl  = (const float*)d_in[6];
  const float* temp = (const float*)d_in[7];

  char* ws = (char*)d_ws;
  float*  g_acc  = (float*)(ws + WS_ACC);
  float*  g_norm = (float*)(ws + WS_NORM);
  __bf16* wcat   = (__bf16*)(ws + WS_WCAT);
  __bf16* wslice = (__bf16*)(ws + WS_WSLICE);

  hipLaunchKernelGGL(k_prep, dim3(256), dim3(256), 0, stream,
                     Wx, Wfx, Wsl, g_acc, g_norm, wcat, wslice);
  hipLaunchKernelGGL(k_main, dim3(256, B_), dim3(256), LDS_BYTES, stream,
                     x, bx, bfx, bsl, temp, wcat, wslice, g_acc, g_norm);
  hipLaunchKernelGGL(k_final, dim3((B_*H_*S_*D_ + 255)/256), dim3(256), 0, stream,
                     g_acc, g_norm, (float*)d_out);
}
